// AttentionABC_87308095193502
// MI455X (gfx1250) — compile-verified, matrix operand order checked by reading, not run on hardware
//
#include <hip/hip_runtime.h>
#include <hip/hip_bf16.h>

typedef __attribute__((ext_vector_type(16))) _Float16 v16h;
typedef __attribute__((ext_vector_type(8)))  _Float16 v8h;
typedef __attribute__((ext_vector_type(8)))  float    v8f;

#define NEGMAX (-3.402823466e38f)

// WMMA 16-bit A/B operand loader (ISA 7.12.2): for a lane in half 'half', v16h elements
// 0..7  = row halves [8*half, 8*half+8)      (VGPRs 0-3: K = 2r+8h, 2r+1+8h)
// 8..15 = row halves [16+8*half, 16+8*half+8) (VGPRs 4-7: K = 16+2(r-4)+8h, ...)
// => two aligned 16-byte loads per lane (global_load_b128 / ds_load_b128).
__device__ __forceinline__ v16h ldop(const _Float16* p, int half) {
    v8h lo = *(const v8h*)(p + 8 * half);
    v8h hi = *(const v8h*)(p + 16 + 8 * half);
    return __builtin_shufflevector(lo, hi, 0, 1, 2, 3, 4, 5, 6, 7,
                                           8, 9, 10, 11, 12, 13, 14, 15);
}

// ---------------- Kernel 0: one-shot precision/layout prep ----------------
// xh  : x converted to f16, row-major [8192][256]
// wqT : w_qkv f16 transposed to [768][256]  (col-major -> contiguous K per output col)
// wpT : w_proj f16 transposed to [256][256]
__global__ __launch_bounds__(256) void prep(
    const float* __restrict__ x, const float* __restrict__ w_qkv,
    const float* __restrict__ w_proj,
    _Float16* __restrict__ xh, _Float16* __restrict__ wqT, _Float16* __restrict__ wpT)
{
    const size_t stride = (size_t)gridDim.x * blockDim.x;
    const size_t i0 = (size_t)blockIdx.x * blockDim.x + threadIdx.x;
    for (size_t t = i0; t < (size_t)8192 * 256; t += stride)
        xh[t] = (_Float16)x[t];
    for (size_t t = i0; t < (size_t)256 * 768; t += stride) {
        const int k = (int)(t / 768), n = (int)(t % 768);
        wqT[(size_t)n * 256 + k] = (_Float16)w_qkv[t];
    }
    for (size_t t = i0; t < (size_t)256 * 256; t += stride) {
        const int k = (int)(t / 256), n = (int)(t % 256);
        wpT[(size_t)n * 256 + k] = (_Float16)w_proj[t];
    }
}

// ---------------- Kernel 1: QKV projection (8192x256 @ 256x768 + bias) ----------------
__global__ __launch_bounds__(256) void qkv_gemm(
    const _Float16* __restrict__ xh, const _Float16* __restrict__ wqT,
    const float* __restrict__ bias,
    _Float16* __restrict__ qc, _Float16* __restrict__ kc, _Float16* __restrict__ vcT)
{
    const int lane = threadIdx.x & 31;
    const int wave = threadIdx.x >> 5;
    const int tile = blockIdx.x * 8 + wave;      // 512 x 48 tiles
    const int tm = tile / 48, tn = tile % 48;
    const int row0 = tm * 16, col0 = tn * 16;
    const int mA = lane & 15, half = lane >> 4;

    const _Float16* arow = xh + (size_t)(row0 + mA) * 256;
    const _Float16* brow = wqT + (size_t)(col0 + mA) * 256;

    v8f c = {};
#pragma unroll
    for (int kk = 0; kk < 8; ++kk) {
        const v16h a = ldop(arow + kk * 32, half);
        const v16h b = ldop(brow + kk * 32, half);
        c = __builtin_amdgcn_wmma_f32_16x16x32_f16(false, a, false, b, (short)0, c, false, false);
    }

    // tsel/h uniform per 16-wide column tile (16 | 256): scalar branch, no divergence
    const int tsel = col0 >> 8;            // 0=q 1=k 2=v
    const int h = (col0 >> 5) & 7;
    const int hd = (col0 & 31) + mA;
    const float bv = bias[col0 + mA];
#pragma unroll
    for (int r = 0; r < 8; ++r) {
        const int row = row0 + r + 8 * half;   // token index in [0,8192)
        const int bi = row >> 11, nn = row & 2047;
        const int bh = bi * 8 + h;
        const _Float16 hv = (_Float16)(c[r] + bv);
        if (tsel == 0)      qc[((size_t)bh * 2048 + nn) * 32 + hd] = hv;
        else if (tsel == 1) kc[((size_t)bh * 2080 + nn) * 32 + hd] = hv;
        else                vcT[((size_t)bh * 32 + hd) * 2080 + nn] = hv;
    }
}

// ---------------- Kernel 1b: memory slots + zero pad (s = 2048..2079) ----------------
__global__ __launch_bounds__(256) void fill_mem(
    const float* __restrict__ mem_k, const float* __restrict__ mem_v,
    _Float16* __restrict__ kc, _Float16* __restrict__ vcT)
{
    const int bh = blockIdx.x;            // 0..31
    const int h = bh & 7;
    for (int t = threadIdx.x; t < 32 * 32; t += blockDim.x) {
        const int srel = t >> 5, hd = t & 31;
        const float kv = (srel < 4) ? mem_k[(h * 4 + srel) * 32 + hd] : 0.f;
        const float vv = (srel < 4) ? mem_v[(h * 4 + srel) * 32 + hd] : 0.f;
        kc[((size_t)bh * 2080 + 2048 + srel) * 32 + hd] = (_Float16)kv;
        vcT[((size_t)bh * 32 + hd) * 2080 + 2048 + srel] = (_Float16)vv;
    }
}

// ---------------- Kernel 2: fused scores -> top-32 threshold -> softmax -> PV ----------------
// One block = one (b,h, 16-row tile). 16 waves. Dynamic LDS (232,448 B < 320 KB WGP LDS):
//   sS : 16 x 2080 f32 scores (133,120 B)
//   sA : 16 x 2080 f16 normalized attention (66,560 B) -> b128 A-operand loads for PV
//   sP : 16 waves x (16x32) f32 PV partials (32,768 B)
__global__ __launch_bounds__(512) void attn_fused(
    const _Float16* __restrict__ qc, const _Float16* __restrict__ kc,
    const _Float16* __restrict__ vcT, const float* __restrict__ scale,
    _Float16* __restrict__ aout)
{
    extern __shared__ char smem[];
    float*    sS = (float*)smem;                              // [16][2080] f32
    _Float16* sA = (_Float16*)(smem + 16 * 2080 * 4);         // [16][2080] f16
    float*    sP = (float*)(smem + 16 * 2080 * 4 + 16 * 2080 * 2);  // [16][512] f32

    const int lane = threadIdx.x & 31;
    const int wave = threadIdx.x >> 5;              // 0..15
    const int blk = blockIdx.x;                     // 0..4095
    const int rt = blk & 127, bh = blk >> 7;
    const int h = bh & 7, b = bh >> 3;
    const int n0 = rt * 16;
    const int mA = lane & 15, half = lane >> 4;
    const float sc = scale[h] * 0.17677669529663687f;   // 32^-0.5

    const _Float16* qbase = qc + ((size_t)bh * 2048 + n0) * 32;
    const _Float16* kbase = kc + (size_t)bh * 2080 * 32;
    const _Float16* vbase = vcT + (size_t)bh * 32 * 2080;

    // A operand: q tile 16x32 (reused by every chunk)
    const v16h aq = ldop(qbase + mA * 32, half);

    // ---- Phase A: scores into LDS (130 chunks of 16 columns) ----
    for (int cs = wave; cs < 130; cs += 16) {
        if (cs + 16 < 130)
            __builtin_prefetch(kbase + (size_t)((cs + 16) * 16 + mA) * 32, 0, 1);
        const v16h bk = ldop(kbase + (size_t)(cs * 16 + mA) * 32, half);
        v8f c = {};
        c = __builtin_amdgcn_wmma_f32_16x16x32_f16(false, aq, false, bk, (short)0, c, false, false);
#pragma unroll
        for (int r = 0; r < 8; ++r)
            sS[(r + 8 * half) * 2080 + cs * 16 + mA] = c[r] * sc;
    }
    __syncthreads();

    // ---- Phase B: per-row mask, exact top-32 threshold, softmax. wave w owns row w ----
    {
        const int m = wave;
        const int grow = n0 + m;                    // diagonal column
        float* rowp = sS + m * 2080;
        _Float16* aprow = sA + m * 2080;
        float v[65];
#pragma unroll
        for (int i = 0; i < 65; ++i) {
            const int col = lane + i * 32;
            float x = rowp[col];
            if (col >= 2052 || col == grow) x = NEGMAX;   // pad + diag mask
            rowp[col] = x;                                 // masked original kept in LDS
            v[i] = x;                                      // working copy in VGPRs
        }
        float vk = NEGMAX, m0 = NEGMAX;
        for (int t = 0; t < 32; ++t) {
            float lm = -__builtin_inff(); int li = 0;
#pragma unroll
            for (int i = 0; i < 65; ++i)
                if (v[i] > lm) { lm = v[i]; li = i; }
            float wm = lm;
#pragma unroll
            for (int off = 16; off; off >>= 1)
                wm = fmaxf(wm, __shfl_xor(wm, off, 32));
            if (t == 0) m0 = wm;                    // row max
            vk = wm;                                // after loop: 32nd largest
            const unsigned long long ball = __ballot(lm == wm);
            const int first = __ffsll(ball) - 1;
            if (lane == first) v[li] = -__builtin_inff();
        }
        // softmax over entries >= vk (others contribute exactly 0, matching exp(-FLT_MAX) underflow)
        float lsum = 0.f;
#pragma unroll
        for (int i = 0; i < 65; ++i) {
            const int col = lane + i * 32;
            const float x = rowp[col];
            const float e = (x >= vk) ? __expf(x - m0) : 0.f;
            rowp[col] = e;
            lsum += e;
        }
#pragma unroll
        for (int off = 16; off; off >>= 1) lsum += __shfl_xor(lsum, off, 32);
        const float inv = 1.f / lsum;
#pragma unroll
        for (int i = 0; i < 65; ++i) {
            const int col = lane + i * 32;
            aprow[col] = (_Float16)(rowp[col] * inv);   // packed f16 attention row
        }
    }
    __syncthreads();

    // ---- Phase C: PV (attn 16x2080 @ v 2080x32), K split across waves, 65 chunks of 32 ----
    v8f c0 = {}, c1 = {};
    for (int cs = wave; cs < 65; cs += 16) {
        const v16h ap = ldop(sA + (size_t)mA * 2080 + cs * 32, half);           // ds_load_b128 x2
        const v16h b0 = ldop(vbase + (size_t)mA * 2080 + cs * 32, half);        // hd 0..15
        const v16h b1 = ldop(vbase + (size_t)(16 + mA) * 2080 + cs * 32, half); // hd 16..31
        c0 = __builtin_amdgcn_wmma_f32_16x16x32_f16(false, ap, false, b0, (short)0, c0, false, false);
        c1 = __builtin_amdgcn_wmma_f32_16x16x32_f16(false, ap, false, b1, (short)0, c1, false, false);
    }
#pragma unroll
    for (int r = 0; r < 8; ++r) {
        const int m = r + 8 * half;
        sP[wave * 512 + m * 32 + mA]      = c0[r];
        sP[wave * 512 + m * 32 + 16 + mA] = c1[r];
    }
    __syncthreads();

    // deterministic serial reduce over the 16 wave partials, emit f16 [b][n][h*32+col]
    {
        const int t = threadIdx.x;                  // 0..511 -> (m,col)
        float acc = 0.f;
#pragma unroll
        for (int w = 0; w < 16; ++w) acc += sP[w * 512 + t];
        const int m = t >> 5, col = t & 31;
        aout[((size_t)b * 2048 + n0 + m) * 256 + h * 32 + col] = (_Float16)acc;
    }
}

// ---------------- Kernel 3: output projection (8192x256 @ 256x256 + bias) -> f32 ----------------
__global__ __launch_bounds__(256) void proj_gemm(
    const _Float16* __restrict__ aout, const _Float16* __restrict__ wpT,
    const float* __restrict__ bias, float* __restrict__ out)
{
    const int lane = threadIdx.x & 31;
    const int wave = threadIdx.x >> 5;
    const int tile = blockIdx.x * 8 + wave;     // 512 x 16 tiles
    const int tm = tile >> 4, tn = tile & 15;
    const int row0 = tm * 16, col0 = tn * 16;
    const int mA = lane & 15, half = lane >> 4;

    const _Float16* arow = aout + (size_t)(row0 + mA) * 256;
    const _Float16* brow = wpT + (size_t)(col0 + mA) * 256;

    v8f c = {};
#pragma unroll
    for (int kk = 0; kk < 8; ++kk) {
        const v16h a = ldop(arow + kk * 32, half);
        const v16h b = ldop(brow + kk * 32, half);
        c = __builtin_amdgcn_wmma_f32_16x16x32_f16(false, a, false, b, (short)0, c, false, false);
    }
    const float bv = bias[col0 + mA];
#pragma unroll
    for (int r = 0; r < 8; ++r)
        out[(size_t)(row0 + r + 8 * half) * 256 + col0 + mA] = c[r] + bv;
}

extern "C" void kernel_launch(void* const* d_in, const int* in_sizes, int n_in,
                              void* d_out, int out_size, void* d_ws, size_t ws_size,
                              hipStream_t stream) {
    const float* x      = (const float*)d_in[0];
    const float* w_qkv  = (const float*)d_in[1];
    const float* b_qkv  = (const float*)d_in[2];
    const float* w_proj = (const float*)d_in[3];
    const float* b_proj = (const float*)d_in[4];
    const float* scale  = (const float*)d_in[5];
    const float* mem_k  = (const float*)d_in[6];
    const float* mem_v  = (const float*)d_in[7];

    size_t off = 0;
    auto carve = [&](size_t bytes) {
        void* p = (char*)d_ws + off;
        off += (bytes + 255) & ~(size_t)255;
        return p;
    };
    _Float16* qc   = (_Float16*)carve((size_t)32 * 2048 * 32 * 2);   // [bh][n][hd]
    _Float16* kc   = (_Float16*)carve((size_t)32 * 2080 * 32 * 2);   // [bh][s_pad][hd]
    _Float16* vcT  = (_Float16*)carve((size_t)32 * 32 * 2080 * 2);   // [bh][hd][s_pad]
    _Float16* aout = (_Float16*)carve((size_t)8192 * 256 * 2);       // [b*n][h*hd]
    _Float16* xh   = (_Float16*)carve((size_t)8192 * 256 * 2);       // x in f16
    _Float16* wqT  = (_Float16*)carve((size_t)768 * 256 * 2);        // w_qkv^T f16
    _Float16* wpT  = (_Float16*)carve((size_t)256 * 256 * 2);        // w_proj^T f16

    prep<<<2048, 256, 0, stream>>>(x, w_qkv, w_proj, xh, wqT, wpT);
    qkv_gemm<<<3072, 256, 0, stream>>>(xh, wqT, b_qkv, qc, kc, vcT);
    fill_mem<<<32, 256, 0, stream>>>(mem_k, mem_v, kc, vcT);

    const size_t smem = (size_t)16 * 2080 * 4 + (size_t)16 * 2080 * 2 + (size_t)16 * 512 * 4;
    attn_fused<<<4096, 512, smem, stream>>>(qc, kc, vcT, scale, aout);

    proj_gemm<<<1024, 256, 0, stream>>>(aout, wpT, b_proj, (float*)d_out);
}
